// GNN_Model_42649025249404
// MI455X (gfx1250) — matrix operand order repaired and checked by reading, not compiled, not run on hardware
//
#include <hip/hip_runtime.h>
#include <hip/hip_bf16.h>
#include <math.h>

// ---------------------------------------------------------------------------
// CDNA5 (gfx1250) bf16 WMMA GEMM pipeline for the GNN propagation model.
// Wave32 only. Matrix op: v_wmma_f32_16x16x32_bf16.
// Tiles staged via GLOBAL_LOAD_ASYNC_TO_LDS_B128 (ASYNCcnt) when available.
// All GEMM operands are pre-converted, zero-padded bf16 => branch-free staging.
// ---------------------------------------------------------------------------

typedef __attribute__((ext_vector_type(16))) __bf16 bf16x16;
typedef __attribute__((ext_vector_type(8)))  __bf16 bf16x8;
typedef __attribute__((ext_vector_type(8)))  float  f32x8;
typedef __attribute__((ext_vector_type(4)))  int    v4i;

#if defined(__has_builtin)
#  if __has_builtin(__builtin_amdgcn_global_load_async_to_lds_b128)
#    define USE_ASYNC_LDS 1
#  endif
#endif
#ifndef USE_ASYNC_LDS
#  define USE_ASYNC_LDS 0
#endif

#define AS1 __attribute__((address_space(1)))
#define AS3 __attribute__((address_space(3)))

union FragBF16 {
    bf16x8  h[2];
    bf16x16 v;
};

static __device__ inline f32x8 wmma_bf16(bf16x16 a, bf16x16 b, f32x8 c) {
    // (neg_a, A, neg_b, B, c_mod, C, reuse_a, reuse_b)
    return __builtin_amdgcn_wmma_f32_16x16x32_bf16(false, a, false, b, (short)0, c,
                                                   false, false);
}

static __device__ inline float sigmoidf(float x) { return 1.0f / (1.0f + expf(-x)); }

static __device__ __forceinline__ void wait_async_lds() {
#if USE_ASYNC_LDS
#  if defined(__has_builtin) && __has_builtin(__builtin_amdgcn_s_wait_asynccnt)
    __builtin_amdgcn_s_wait_asynccnt(0);
#  else
    asm volatile("s_wait_asynccnt 0x0" ::: "memory");
#  endif
#endif
}

#define NN 128            // nodes
#define NSQ 16384         // N*N

// ---------------------------------------------------------------------------
// Stage a [ROWS x KCHUNKS*8] bf16 tile into LDS. Fully unrolled, branch-free:
// ROWS*KCHUNKS is a multiple of 256 at every call site, and all sources are
// zero-padded so every 16B chunk is in bounds.
// ---------------------------------------------------------------------------
template<int ROWS, int KCHUNKS, int DSTSTRIDE>
static __device__ __forceinline__
void stage_lds(__bf16* __restrict__ dst,
               const __bf16* __restrict__ src, int srcStride,
               int rowBase, int k0, int tid) {
    constexpr int ITER = (ROWS * KCHUNKS) / 256;
    #pragma unroll
    for (int it = 0; it < ITER; ++it) {
        const int s = tid + it * 256;
        const int r = s / KCHUNKS;
        const int c = (s % KCHUNKS) * 8;
        const __bf16* p = src + (size_t)(rowBase + r) * srcStride + k0 + c;
        __bf16* q = dst + r * DSTSTRIDE + c;
#if USE_ASYNC_LDS
        __builtin_amdgcn_global_load_async_to_lds_b128((AS1 v4i*)p, (AS3 v4i*)q,
                                                       0, 0);
#else
        *(bf16x8*)q = *(const bf16x8*)p;
#endif
    }
}

// ---------------------------------------------------------------------------
// GEMM: Out[M, Nc] = epilogue( A[M,K] @ W[Nc_pad,K]^T + bias )
//  A, W bf16 row-major, K multiple of 64, M multiple of 64, W rows padded so
//  every staged tile is fully resident. Epilogue options:
//  - bias fp32 (optional)
//  - rowAdd (optional): += rowAdd[(b*128 + w)*Nc + col], row = (b,i,w) packed
//  - relu / gate (optional): relu, then *= gate[row] (sigmoid adjacency)
//  - transStore: store row' = (b,w,i)  (fused edge transpose)
//  - outBF16: emit bf16 (tensor feeds another GEMM) else fp32
// Tile: BM=64, BN=64, BK=64. 256 threads = 8 waves; 4 WMMA per staged chunk.
// ---------------------------------------------------------------------------
__global__ __launch_bounds__(256)
void gemm_bf16_wmma(const __bf16* __restrict__ A,
                    const __bf16* __restrict__ W,
                    const float* __restrict__ bias,
                    void* __restrict__ Out,
                    int M, int Nc, int K,
                    int relu,
                    const float* __restrict__ rowAdd,
                    const float* __restrict__ gate,
                    int transStore, int outBF16) {
    __shared__ __bf16 As[64][72];   // rows 144B (16B aligned), +8 pad
    __shared__ __bf16 Bs[64][72];

    const int tid  = threadIdx.x;
    const int wid  = tid >> 5;          // 0..7
    const int lane = tid & 31;
    const int half = lane >> 4;         // 0/1
    const int r16  = lane & 15;

    const int m0 = blockIdx.x * 64;
    const int n0 = blockIdx.y * 64;

    const int mi  = wid & 3;            // M sub-tile 0..3
    const int ni0 = wid >> 2;           // N sub-tiles ni0, ni0+2

    f32x8 acc0 = {0.f,0.f,0.f,0.f,0.f,0.f,0.f,0.f};
    f32x8 acc1 = {0.f,0.f,0.f,0.f,0.f,0.f,0.f,0.f};

    for (int k0 = 0; k0 < K; k0 += 64) {
        __syncthreads();
        stage_lds<64, 8, 72>(&As[0][0], A, K, m0, k0, tid);
        stage_lds<64, 8, 72>(&Bs[0][0], W, K, n0, k0, tid);
        wait_async_lds();
        __syncthreads();

        #pragma unroll
        for (int kh = 0; kh < 2; ++kh) {
            // A fragment (16x32 MxK): lane=(half,r16) -> M=r16
            FragBF16 af;
            {
                const __bf16* p = &As[mi * 16 + r16][kh * 32 + half * 16];
                af.h[0] = *(const bf16x8*)(p);        // K = half*16 + 0..7
                af.h[1] = *(const bf16x8*)(p + 8);    // K = half*16 + 8..15
            }
            // B fragments (32x16 KxN): lane=(half,r16) -> N=r16, K=half*16+j
            FragBF16 bf0, bf1;
            {
                const __bf16* p = &Bs[ni0 * 16 + r16][kh * 32 + half * 16];
                bf0.h[0] = *(const bf16x8*)(p);
                bf0.h[1] = *(const bf16x8*)(p + 8);
            }
            {
                const __bf16* p = &Bs[(ni0 + 2) * 16 + r16][kh * 32 + half * 16];
                bf1.h[0] = *(const bf16x8*)(p);
                bf1.h[1] = *(const bf16x8*)(p + 8);
            }
            acc0 = wmma_bf16(af.v, bf0.v, acc0);
            acc1 = wmma_bf16(af.v, bf1.v, acc1);
        }
    }

    // --- epilogue: D layout M = v + 8*half, N = r16 ---
    #pragma unroll
    for (int t = 0; t < 2; ++t) {
        const f32x8 acc = t ? acc1 : acc0;
        const int ni  = ni0 + t * 2;
        const int col = n0 + ni * 16 + r16;
        if (col < Nc) {
            #pragma unroll
            for (int v = 0; v < 8; ++v) {
                const int row = m0 + mi * 16 + v + 8 * half;
                float val = acc[v];
                if (bias)   val += bias[col];
                if (rowAdd) {
                    const int b = row >> 14;            // row = b*N*N + i*N + w
                    const int w = row & (NN - 1);
                    val += rowAdd[(size_t)(b * NN + w) * Nc + col];
                }
                if (relu) val = fmaxf(val, 0.f);
                if (gate) val *= gate[row];
                size_t orow = (size_t)row;
                if (transStore) {
                    const int b   = row >> 14;
                    const int rem = row & (NSQ - 1);
                    const int i   = rem >> 7;
                    const int w   = rem & (NN - 1);
                    orow = (size_t)b * NSQ + (size_t)w * NN + i;
                }
                const size_t idx = orow * (size_t)Nc + col;
                if (outBF16) ((__bf16*)Out)[idx] = (__bf16)val;
                else         ((float*)Out)[idx]  = val;
            }
        }
    }
}

// ---------------------------------------------------------------------------
// Fused link-function tail:  pred = relu(X1 @ Wl2^T + bl2) @ Wl3^T + bl3
// sig = sigmoid(pred).  Avoids materializing the 268MB X2 tensor.
// Block: 64 edge rows, 8 waves: (mi = wid&3) M sub-tile, (wid>>2) selects half
// of the 512 hidden channels (16 n-tiles each), K=512 staged at BK=32.
// ---------------------------------------------------------------------------
__global__ __launch_bounds__(256)
void link23_kernel(const __bf16* __restrict__ X1,
                   const __bf16* __restrict__ Wl2, const float* __restrict__ bl2,
                   const float* __restrict__ w3,  const float* __restrict__ bl3,
                   float* __restrict__ pred, float* __restrict__ sig) {
    __shared__ __bf16 As[64][40];
    __shared__ __bf16 Bs[512][40];
    __shared__ float  red[4][16][2];

    const int tid  = threadIdx.x;
    const int wid  = tid >> 5;
    const int lane = tid & 31;
    const int half = lane >> 4;
    const int r16  = lane & 15;
    const int mi   = wid & 3;
    const int nhalf = wid >> 2;          // 0/1 -> channels [nhalf*256, +256)
    const int m0   = blockIdx.x * 64;

    f32x8 acc[16];
    #pragma unroll
    for (int t = 0; t < 16; ++t) acc[t] = (f32x8){0.f,0.f,0.f,0.f,0.f,0.f,0.f,0.f};

    for (int k0 = 0; k0 < 512; k0 += 32) {
        __syncthreads();
        stage_lds<64,  4, 40>(&As[0][0], X1,  512, m0, k0, tid);
        stage_lds<512, 4, 40>(&Bs[0][0], Wl2, 512, 0,  k0, tid);
        wait_async_lds();
        __syncthreads();

        FragBF16 af;
        {
            const __bf16* p = &As[mi * 16 + r16][half * 16];
            af.h[0] = *(const bf16x8*)(p);
            af.h[1] = *(const bf16x8*)(p + 8);
        }
        #pragma unroll
        for (int t = 0; t < 16; ++t) {
            FragBF16 bfr;
            const __bf16* p = &Bs[nhalf * 256 + t * 16 + r16][half * 16];
            bfr.h[0] = *(const bf16x8*)(p);
            bfr.h[1] = *(const bf16x8*)(p + 8);
            acc[t] = wmma_bf16(af.v, bfr.v, acc[t]);
        }
    }

    // epilogue: relu(acc + bl2[n]) * w3[n], reduce over n
    float rowsum[8];
    #pragma unroll
    for (int v = 0; v < 8; ++v) rowsum[v] = 0.f;
    #pragma unroll
    for (int t = 0; t < 16; ++t) {
        const int n = nhalf * 256 + t * 16 + r16;
        const float b2 = bl2[n];
        const float g3 = w3[n];
        #pragma unroll
        for (int v = 0; v < 8; ++v)
            rowsum[v] += fmaxf(acc[t][v] + b2, 0.f) * g3;
    }
    // cross-lane reduce within each 16-lane group (xor < 16 stays in group)
    #pragma unroll
    for (int v = 0; v < 8; ++v) {
        float s = rowsum[v];
        s += __shfl_xor(s, 8);
        s += __shfl_xor(s, 4);
        s += __shfl_xor(s, 2);
        s += __shfl_xor(s, 1);
        rowsum[v] = s;
    }
    if (r16 == 0) {
        #pragma unroll
        for (int v = 0; v < 8; ++v)
            red[mi][half * 8 + v][nhalf] = rowsum[v];
    }
    __syncthreads();
    if (tid < 64) {
        const float s = red[tid >> 4][tid & 15][0] + red[tid >> 4][tid & 15][1] + bl3[0];
        const int g = m0 + tid;
        pred[g] = s;
        sig[g]  = sigmoidf(s);
    }
}

// fp32 [R,C] -> bf16 [*, Cpad], zero padded (rows beyond R and cols beyond C)
__global__ __launch_bounds__(256)
void cvt_pad_kernel(const float* __restrict__ src, int srcStride, int R, int C,
                    __bf16* __restrict__ dst, int Cpad, int total) {
    const int idx = blockIdx.x * blockDim.x + threadIdx.x;
    if (idx >= total) return;
    const int r = idx / Cpad;
    const int c = idx % Cpad;
    const float v = (r < R && c < C) ? src[(size_t)r * srcStride + c] : 0.f;
    dst[idx] = (__bf16)v;
}

// m[b,i,:] = sum_w estate[(b,w,i),:]   (estate stores transposed messages)
__global__ __launch_bounds__(256)
void sum_sources_kernel(const __bf16* __restrict__ est, __bf16* __restrict__ m) {
    const int idx = blockIdx.x * blockDim.x + threadIdx.x;   // [1024*256)
    if (idx >= 1024 * 256) return;
    const int bi = idx >> 8;            // b*128 + i
    const int n  = idx & 255;
    const int b  = bi >> 7;
    const int i  = bi & 127;
    const __bf16* p = est + ((size_t)b * NSQ + i) * 256 + n;
    float s = 0.f;
    #pragma unroll 4
    for (int w = 0; w < NN; ++w) s += (float)p[(size_t)w * NN * 256];
    m[idx] = (__bf16)s;
}

// GRU single step: h' = (1-z)*n + z*h
__global__ __launch_bounds__(256)
void gru_kernel(const float* __restrict__ gi, const float* __restrict__ gh,
                const __bf16* __restrict__ h, __bf16* __restrict__ hn) {
    const int idx = blockIdx.x * blockDim.x + threadIdx.x;
    if (idx >= 1024 * 256) return;
    const int row = idx >> 8;
    const int d0  = idx & 255;
    const size_t base = (size_t)row * 768;
    const float r = sigmoidf(gi[base + d0]       + gh[base + d0]);
    const float z = sigmoidf(gi[base + 256 + d0] + gh[base + 256 + d0]);
    const float n = tanhf   (gi[base + 512 + d0] + r * gh[base + 512 + d0]);
    const float hp = (float)h[idx];
    hn[idx] = (__bf16)((1.f - z) * n + z * hp);
}

// ---------------------------------------------------------------------------
extern "C" void kernel_launch(void* const* d_in, const int* in_sizes, int n_in,
                              void* d_out, int out_size, void* d_ws, size_t ws_size,
                              hipStream_t stream) {
    const float* edge_f = (const float*)d_in[0];   // [8,128,128,200]
    const float* node_f = (const float*)d_in[1];   // [8,128,100]
    const float* W_er = (const float*)d_in[2];  const float* b_er = (const float*)d_in[3];
    const float* W_nr = (const float*)d_in[4];  const float* b_nr = (const float*)d_in[5];
    const float* Wl1  = (const float*)d_in[6];  const float* bl1  = (const float*)d_in[7];
    const float* Wl2  = (const float*)d_in[8];  const float* bl2  = (const float*)d_in[9];
    const float* Wl3  = (const float*)d_in[10]; const float* bl3  = (const float*)d_in[11];
    const float* W_m  = (const float*)d_in[12]; const float* b_m  = (const float*)d_in[13];
    const float* W_ih = (const float*)d_in[14]; const float* b_ih = (const float*)d_in[15];
    const float* W_hh = (const float*)d_in[16]; const float* b_hh = (const float*)d_in[17];
    const float* W_r  = (const float*)d_in[18]; const float* b_r  = (const float*)d_in[19];

    const int ME = 8 * 128 * 128;   // 131072 edge rows

    // byte-granular workspace carve, 256B aligned
    char* base = (char*)d_ws;
    size_t off = 0;
    auto alloc = [&](size_t bytes) -> void* {
        void* p = base + off;
        off = (off + bytes + 255) & ~(size_t)255;
        return p;
    };
    __bf16* edge_b = (__bf16*)alloc((size_t)ME * 256 * 2);   // padded 200->256
    __bf16* node_b = (__bf16*)alloc((size_t)1024 * 128 * 2); // padded 100->128
    __bf16* ef     = (__bf16*)alloc((size_t)ME * 256 * 2);   // resized edge feats
    __bf16* eA     = (__bf16*)alloc((size_t)ME * 256 * 2);   // e_state ping
    __bf16* eB     = (__bf16*)alloc((size_t)ME * 256 * 2);   // e_state pong
    __bf16* X1     = (__bf16*)alloc((size_t)ME * 512 * 2);   // link hidden 1
    float*  sg     = (float*) alloc((size_t)ME * 4);         // sigmoid(adj)
    float*  Hm     = (float*) alloc((size_t)1024 * 256 * 4); // h_w@Wm1^T + b_m
    __bf16* hA     = (__bf16*)alloc((size_t)1024 * 256 * 2);
    __bf16* hB     = (__bf16*)alloc((size_t)1024 * 256 * 2);
    __bf16* mB     = (__bf16*)alloc((size_t)1024 * 256 * 2);
    float*  gi     = (float*) alloc((size_t)1024 * 768 * 4);
    float*  gh     = (float*) alloc((size_t)1024 * 768 * 4);
    // padded bf16 weights
    __bf16* Wb_er = (__bf16*)alloc((size_t)256 * 256 * 2);
    __bf16* Wb_nr = (__bf16*)alloc((size_t)256 * 128 * 2);
    __bf16* Wb_l1 = (__bf16*)alloc((size_t)512 * 256 * 2);
    __bf16* Wb_l2 = (__bf16*)alloc((size_t)512 * 512 * 2);
    __bf16* Wb_m1 = (__bf16*)alloc((size_t)256 * 256 * 2);
    __bf16* Wb_m2 = (__bf16*)alloc((size_t)256 * 256 * 2);
    __bf16* Wb_ih = (__bf16*)alloc((size_t)768 * 256 * 2);
    __bf16* Wb_hh = (__bf16*)alloc((size_t)768 * 256 * 2);
    __bf16* Wb_r  = (__bf16*)alloc((size_t)128 * 256 * 2);   // 117 -> 128 rows

    float* pred   = (float*)d_out;            // [8,128,128]
    float* labels = (float*)d_out + ME;       // [8,128,117]

    dim3 blk(256);
    auto cvt = [&](const float* s, int ss, int R, int C, __bf16* d, int Cp, int tot) {
        cvt_pad_kernel<<<(tot + 255) / 256, blk, 0, stream>>>(s, ss, R, C, d, Cp, tot);
    };
    auto gemm = [&](const __bf16* A, const __bf16* W, const float* bias, void* Out,
                    int M, int Nc, int K, int relu,
                    const float* rowAdd, const float* gate, int trans, int obf) {
        dim3 grid((M + 63) / 64, (Nc + 63) / 64);
        gemm_bf16_wmma<<<grid, blk, 0, stream>>>(A, W, bias, Out, M, Nc, K,
                                                 relu, rowAdd, gate, trans, obf);
    };

    // -------- one-time bf16 conversion / padding --------
    cvt(edge_f, 200, ME, 200, edge_b, 256, ME * 256);
    cvt(node_f, 100, 1024, 100, node_b, 128, 1024 * 128);
    cvt(W_er, 200, 256, 200, Wb_er, 256, 256 * 256);
    cvt(W_nr, 100, 256, 100, Wb_nr, 128, 256 * 128);
    cvt(Wl1, 256, 512, 256, Wb_l1, 256, 512 * 256);
    cvt(Wl2, 512, 512, 512, Wb_l2, 512, 512 * 512);
    cvt(W_m, 512, 256, 256, Wb_m1, 256, 256 * 256);          // cols [0,256)
    cvt(W_m + 256, 512, 256, 256, Wb_m2, 256, 256 * 256);    // cols [256,512)
    cvt(W_ih, 256, 768, 256, Wb_ih, 256, 768 * 256);
    cvt(W_hh, 256, 768, 256, Wb_hh, 256, 768 * 256);
    cvt(W_r, 256, 117, 256, Wb_r, 256, 128 * 256);

    // -------- feature resize --------
    gemm(edge_b, Wb_er, b_er, ef, ME,   256, 256, 0, nullptr, nullptr, 0, 1);
    gemm(node_b, Wb_nr, b_nr, hA, 1024, 256, 128, 0, nullptr, nullptr, 0, 1);

    const __bf16* estate = ef;    // initial e_state
    __bf16* hc = hA; __bf16* hn = hB;
    __bf16* enext = eA;

    for (int layer = 0; layer < 3; ++layer) {
        // link function: X1 = relu(e_state @ Wl1^T + bl1); fused Wl2/Wl3 tail
        gemm(estate, Wb_l1, bl1, X1, ME, 512, 256, 1, nullptr, nullptr, 0, 1);
        link23_kernel<<<ME / 64, blk, 0, stream>>>(X1, Wb_l2, bl2, Wl3, bl3, pred, sg);

        // Hm[b*128+w,:] = h[b,w] @ W_m[:, :256]^T + b_m   (broadcast concat half)
        gemm(hc, Wb_m1, b_m, Hm, 1024, 256, 256, 0, nullptr, nullptr, 0, 0);

        // M = sig * relu(ef @ W_m[:,256:]^T + Hm); transposed store -> new e_state
        gemm(ef, Wb_m2, nullptr, enext, ME, 256, 256, 1, Hm, sg, 1, 1);

        // aggregate messages over sources, then GRU update
        sum_sources_kernel<<<(1024 * 256 + 255) / 256, blk, 0, stream>>>(enext, mB);
        gemm(mB, Wb_ih, b_ih, gi, 1024, 768, 256, 0, nullptr, nullptr, 0, 0);
        gemm(hc, Wb_hh, b_hh, gh, 1024, 768, 256, 0, nullptr, nullptr, 0, 0);
        gru_kernel<<<(1024 * 256 + 255) / 256, blk, 0, stream>>>(gi, gh, hc, hn);

        // rotate buffers
        __bf16* t = hc; hc = hn; hn = t;
        estate = enext;
        enext  = (enext == eA) ? eB : eA;
    }

    // readout: labels = h @ W_r^T + b_r   [1024 x 117], padded weight rows
    gemm(hc, Wb_r, b_r, labels, 1024, 117, 256, 0, nullptr, nullptr, 0, 0);

    (void)in_sizes; (void)n_in; (void)out_size; (void)ws_size;
}
